// Decoder_62869731279458
// MI455X (gfx1250) — compile-verified, matrix operand order checked by reading, not run on hardware
//
#include <hip/hip_runtime.h>

typedef __attribute__((ext_vector_type(16))) _Float16 v16h;
typedef __attribute__((ext_vector_type(8)))  float    v8f;

#define BQ 128
#define TQ 64
#define HQ 128

// ---------- fast transcendentals (CDNA5 has V_TANH_F32) ----------
__device__ __forceinline__ float fast_tanh(float x) {
#if __has_builtin(__builtin_amdgcn_tanhf)
  return __builtin_amdgcn_tanhf(x);
#elif __has_builtin(__builtin_amdgcn_tanh_f32)
  return __builtin_amdgcn_tanh_f32(x);
#else
  return tanhf(x);
#endif
}
__device__ __forceinline__ float fast_sigmoid(float x) {
  return 0.5f * fast_tanh(0.5f * x) + 0.5f;
}

// ---------- WMMA fragment helpers (wave32, v_wmma_f32_16x16x32_f16) ----------
// A tile 16x32 (ISA 7.12.2): lane m = lane&15; lanes<16 hold K 0..7 & 16..23,
// lanes>=16 hold K 8..15 & 24..31.
__device__ __forceinline__ v16h frag_a_from_f32(const float* __restrict__ base,
                                                int ld, int m0, int k0, int lane) {
  const float* row = base + (size_t)(m0 + (lane & 15)) * ld + k0 + ((lane >> 4) << 3);
  v16h a;
#pragma unroll
  for (int e = 0; e < 8; ++e) {
    a[e]     = (_Float16)row[e];
    a[e + 8] = (_Float16)row[e + 16];
  }
  return a;
}

// B tile 32x16 with B[k][n] = W[(n0+n)*ld + k0+k]  -> computes X @ W^T for row-major W.
__device__ __forceinline__ v16h frag_bT_from_f32(const float* __restrict__ W,
                                                 int ld, int n0, int k0, int lane) {
  const float* col = W + (size_t)(n0 + (lane & 15)) * ld + k0 + ((lane >> 4) << 3);
  v16h b;
#pragma unroll
  for (int e = 0; e < 8; ++e) {
    b[e]     = (_Float16)col[e];
    b[e + 8] = (_Float16)col[e + 16];
  }
  return b;
}

// f16-direct fragment loaders (no conversions; two contiguous 16B chunks each).
// Address space (LDS vs global) is inferred per call site after inlining.
__device__ __forceinline__ v16h frag_a_from_f16(const _Float16* base,
                                                int ld, int m0, int k0, int lane) {
  const _Float16* row = base + (size_t)(m0 + (lane & 15)) * ld + k0 + ((lane >> 4) << 3);
  v16h a;
#pragma unroll
  for (int e = 0; e < 8; ++e) {
    a[e]     = row[e];
    a[e + 8] = row[e + 16];
  }
  return a;
}

__device__ __forceinline__ v16h frag_bT_from_f16(const _Float16* W,
                                                 int ld, int n0, int k0, int lane) {
  const _Float16* col = W + (size_t)(n0 + (lane & 15)) * ld + k0 + ((lane >> 4) << 3);
  v16h b;
#pragma unroll
  for (int e = 0; e < 8; ++e) {
    b[e]     = col[e];
    b[e + 8] = col[e + 16];
  }
  return b;
}

// C/D 16x16 f32: lane n = lane&15, rows = 8*(lane>>4) + v.
__device__ __forceinline__ void store_c_f32(float* __restrict__ dst, int ld,
                                            int m0, int n0, int lane, v8f c) {
  float* p = dst + (size_t)(m0 + ((lane >> 4) << 3)) * ld + n0 + (lane & 15);
#pragma unroll
  for (int v = 0; v < 8; ++v) p[(size_t)v * ld] = c[v];
}

// Same, plus a per-column bias (bias indexed by n).
__device__ __forceinline__ void store_c_f32_bias(float* __restrict__ dst, int ld,
                                                 int m0, int n0, int lane, v8f c,
                                                 const float* __restrict__ bias) {
  int n = n0 + (lane & 15);
  float bv = bias[n];
  float* p = dst + (size_t)(m0 + ((lane >> 4) << 3)) * ld + n;
#pragma unroll
  for (int v = 0; v < 8; ++v) p[(size_t)v * ld] = c[v] + bv;
}

__device__ __forceinline__ void store_c_f16(_Float16* dst, int ld,
                                            int m0, int n0, int lane, v8f c) {
  _Float16* p = dst + (size_t)(m0 + ((lane >> 4) << 3)) * ld + n0 + (lane & 15);
#pragma unroll
  for (int v = 0; v < 8; ++v) p[(size_t)v * ld] = (_Float16)c[v];
}

// ---------- kernel 0: zero LSTM state ----------
__global__ void init_state_kernel(float* __restrict__ h, float* __restrict__ cc,
                                  _Float16* __restrict__ cc16) {
  int i = blockIdx.x * blockDim.x + threadIdx.x;
  if (i < BQ * HQ) {
    h[i] = 0.0f; cc[i] = 0.0f; cc16[i] = (_Float16)0.0f;
  }
}

// ---------- kernel 1: preEnc[b*T+t, j] = X_encoded[b,t,:] . W1[j, 256:384] + b1[j] ----------
// M = B*T = 8192 rows, N = 128, K = 128. One 16-row M-tile per block, one N-tile per wave.
// b1 is folded in here so the per-step attention loop never touches it.
__global__ void __launch_bounds__(256)
precompute_enc_kernel(const float* __restrict__ Xe, const float* __restrict__ W1,
                      const float* __restrict__ b1, float* __restrict__ preEnc) {
  int lane = threadIdx.x & 31;
  int wid  = threadIdx.x >> 5;      // 0..7 -> N tile
  int m0   = blockIdx.x * 16;       // 512 blocks
  v8f c = {};
#pragma unroll
  for (int kk = 0; kk < 4; ++kk) {
    v16h a = frag_a_from_f32(Xe, HQ, m0, kk * 32, lane);
    v16h b = frag_bT_from_f32(W1 + 256, 3 * HQ, wid * 16, kk * 32, lane);
    c = __builtin_amdgcn_wmma_f32_16x16x32_f16(false, a, false, b, (short)0, c, false, false);
  }
  store_c_f32_bias(preEnc, HQ, m0, wid * 16, lane, c, b1);
}

// ---------- kernel 2: persistent single-workgroup 64-step scan ----------
// WMMA B-operands (W1 h|cc slab, Whh) and the h16 A-operand live in LDS.
__global__ void __launch_bounds__(1024)
decoder_scan_kernel(const float* __restrict__ Xe, const float* __restrict__ Xt,
                    const float* __restrict__ W1,
                    const float* __restrict__ W2, const float* __restrict__ b2,
                    const float* __restrict__ Wfc, const float* __restrict__ bfc,
                    const float* __restrict__ Wih, const float* __restrict__ bih,
                    const float* __restrict__ Whh, const float* __restrict__ bhh,
                    const float* __restrict__ Wfin, const float* __restrict__ bfin,
                    const float* __restrict__ preEnc,
                    float* __restrict__ h_g, float* __restrict__ cc_g,
                    _Float16* __restrict__ cc16_g,
                    float* __restrict__ ctx_g, float* __restrict__ gates_g,
                    float* __restrict__ out) {
  __shared__ _Float16 w1h_s[HQ * 256];      // 64 KB  : W1[:, 0:256] as f16
  __shared__ _Float16 whh_s[4 * HQ * HQ];   // 128 KB : Whh as f16
  __shared__ _Float16 h16_s[BQ * HQ];       // 32 KB  : hidden state (WMMA A operand)
  __shared__ _Float16 s_s[BQ * HQ];         // 32 KB  : per-step attention pre-activation
  __shared__ float a_s[BQ * TQ];            // 32 KB  : logits -> softmax weights
  __shared__ float ypart_s[8 * HQ];         // 4 KB   : phase-5 partial sums
  __shared__ float w2_s[HQ];
  __shared__ float bias_s[4 * HQ], wih_s[4 * HQ];
  __shared__ float wfc_s[2 * BQ];
  __shared__ float y_s[BQ];
  __shared__ float scal_s[2];

  const int tid  = threadIdx.x;
  const int lane = tid & 31;
  const int wid  = tid >> 5;         // 0..31 (wave32)

  // One-shot prologue: stage f16 weights into LDS, zero h16, small consts.
  for (int i = tid; i < HQ * 256; i += 1024) {
    int j = i >> 8, k = i & 255;
    w1h_s[i] = (_Float16)W1[(size_t)j * (3 * HQ) + k];
  }
  for (int i = tid; i < 4 * HQ * HQ; i += 1024) whh_s[i] = (_Float16)Whh[i];
  for (int i = tid; i < BQ * HQ; i += 1024)     h16_s[i] = (_Float16)0.0f;
  for (int i = tid; i < HQ; i += 1024)     w2_s[i] = W2[i];
  for (int i = tid; i < 4 * HQ; i += 1024) { bias_s[i] = bih[i] + bhh[i]; wih_s[i] = Wih[i]; }
  for (int i = tid; i < 2 * BQ; i += 1024) wfc_s[i] = Wfc[i];
  if (tid == 0) { scal_s[0] = b2[0]; scal_s[1] = bfc[0]; }
  __syncthreads();

  for (int t = 0; t < TQ; ++t) {
    // Phase 1: s = [h | cc] @ W1[:, 0:256]^T   (M=128,N=128,K=256) -> f16 in LDS
    for (int ti = wid; ti < 64; ti += 32) {
      int mt = ti >> 3, nt = ti & 7;
      v8f c = {};
#pragma unroll
      for (int kk = 0; kk < 4; ++kk) {       // K 0..127 : h (LDS)
        int k0 = kk * 32;
        v16h a  = frag_a_from_f16(h16_s, HQ, mt * 16, k0, lane);
        v16h bm = frag_bT_from_f16(w1h_s, 2 * HQ, nt * 16, k0, lane);
        c = __builtin_amdgcn_wmma_f32_16x16x32_f16(false, a, false, bm, (short)0, c, false, false);
      }
#pragma unroll
      for (int kk = 0; kk < 4; ++kk) {       // K 128..255 : cc (global)
        int k0 = kk * 32;
        v16h a  = frag_a_from_f16(cc16_g, HQ, mt * 16, k0, lane);
        v16h bm = frag_bT_from_f16(w1h_s, 2 * HQ, nt * 16, HQ + k0, lane);
        c = __builtin_amdgcn_wmma_f32_16x16x32_f16(false, a, false, bm, (short)0, c, false, false);
      }
      store_c_f16(s_s, HQ, mt * 16, nt * 16, lane, c);
    }
    __syncthreads();

    // Phase 2: logits a[b,tt] = tanh(s[b,:] + preEnc'[b,tt,:]) . W2 + b2   (b1 prefolded)
    for (int p = tid; p < BQ * TQ; p += 1024) {
      int b = p >> 6;
      const _Float16* srow = s_s + (size_t)b * HQ;
      const float* pe = preEnc + (size_t)p * HQ;
      __builtin_prefetch(preEnc + (size_t)(p + 1024) * HQ, 0, 0);  // global_prefetch_b8
      float acc = scal_s[0];
#pragma unroll 4
      for (int j = 0; j < HQ; ++j) {
        float v = (float)srow[j] + pe[j];
        acc += fast_tanh(v) * w2_s[j];
      }
      a_s[p] = acc;
    }
    __syncthreads();

    // Phase 3: softmax over tt (row per thread)
    if (tid < BQ) {
      float* row = a_s + tid * TQ;
      float mx = row[0];
      for (int i = 1; i < TQ; ++i) mx = fmaxf(mx, row[i]);
      float sum = 0.0f;
      for (int i = 0; i < TQ; ++i) { float e = __expf(row[i] - mx); row[i] = e; sum += e; }
      float inv = 1.0f / sum;
      for (int i = 0; i < TQ; ++i) row[i] *= inv;
    }
    __syncthreads();

    // Phase 4: context[b,e] = sum_tt beta[b,tt] * Xe[b,tt,e]  (float4-vectorized)
    for (int p = tid; p < BQ * (HQ / 4); p += 1024) {
      int b = p >> 5, e4 = (p & 31) << 2;
      const float4* xe = (const float4*)(Xe + ((size_t)b * TQ) * HQ + e4);
      const float* beta = a_s + b * TQ;
      float4 acc = make_float4(0.f, 0.f, 0.f, 0.f);
#pragma unroll 4
      for (int tt = 0; tt < TQ; ++tt) {
        float4 x = xe[(size_t)tt * (HQ / 4)];
        float w = beta[tt];
        acc.x += w * x.x; acc.y += w * x.y; acc.z += w * x.z; acc.w += w * x.w;
      }
      *(float4*)(ctx_g + (size_t)b * HQ + e4) = acc;
    }
    __threadfence_block(); __syncthreads();   // same-WGP consumers: workgroup scope is enough

    // Phase 5: y[e] = Wfc[0,:128].ctx[:,e] + Wfc[0,128:].Xt[:,t,e] + bfc
    // All 1024 threads: 8 bb-chunks x 128 e, then LDS reduce.
    {
      int e = tid & (HQ - 1), chunk = tid >> 7;
      float acc = 0.0f;
#pragma unroll 4
      for (int q = 0; q < 16; ++q) {
        int bb = chunk * 16 + q;
        acc += wfc_s[bb] * ctx_g[(size_t)bb * HQ + e];
        acc += wfc_s[BQ + bb] * Xt[((size_t)bb * TQ + t) * HQ + e];
      }
      ypart_s[chunk * HQ + e] = acc;
    }
    __syncthreads();
    if (tid < HQ) {
      float acc = scal_s[1];
#pragma unroll
      for (int q = 0; q < 8; ++q) acc += ypart_s[q * HQ + tid];
      y_s[tid] = acc;
    }
    __syncthreads();

    // Phase 6: gates_hh = h @ Whh^T   (M=128, N=512, K=128), A and B from LDS
    for (int ti = wid; ti < 256; ti += 32) {
      int mt = ti >> 5, nt = ti & 31;
      v8f c = {};
#pragma unroll
      for (int kk = 0; kk < 4; ++kk) {
        v16h a  = frag_a_from_f16(h16_s, HQ, mt * 16, kk * 32, lane);
        v16h bm = frag_bT_from_f16(whh_s, HQ, nt * 16, kk * 32, lane);
        c = __builtin_amdgcn_wmma_f32_16x16x32_f16(false, a, false, bm, (short)0, c, false, false);
      }
      store_c_f32(gates_g, 4 * HQ, mt * 16, nt * 16, lane, c);
    }
    __threadfence_block(); __syncthreads();

    // Phase 7: LSTM cell update (rank-1 y*Wih + biases folded in here).
    // fp32 recurrence state; f16 mirrors for next step's WMMA A-operands.
    for (int p = tid; p < BQ * HQ; p += 1024) {
      int b = p >> 7, u = p & (HQ - 1);
      float yb = y_s[b];
      const float* gr = gates_g + (size_t)b * 4 * HQ;
      float gi = gr[u]            + yb * wih_s[u]            + bias_s[u];
      float gf = gr[HQ + u]       + yb * wih_s[HQ + u]       + bias_s[HQ + u];
      float gg = gr[2 * HQ + u]   + yb * wih_s[2 * HQ + u]   + bias_s[2 * HQ + u];
      float go = gr[3 * HQ + u]   + yb * wih_s[3 * HQ + u]   + bias_s[3 * HQ + u];
      float c2 = fast_sigmoid(gf) * cc_g[p] + fast_sigmoid(gi) * fast_tanh(gg);
      float h2 = fast_sigmoid(go) * fast_tanh(c2);
      cc_g[p] = c2;  cc16_g[p] = (_Float16)c2;
      h_g[p]  = h2;  h16_s[p]  = (_Float16)h2;
    }
    __threadfence_block(); __syncthreads();
  }

  // Final: out[b] = [h, ctx] . Wfin + bfin
  if (tid < BQ) {
    int b = tid;
    float acc = bfin[0];
    for (int u = 0; u < HQ; ++u) acc += h_g[(size_t)b * HQ + u] * Wfin[u];
    for (int e = 0; e < HQ; ++e) acc += ctx_g[(size_t)b * HQ + e] * Wfin[HQ + e];
    out[b] = acc;
  }
}

extern "C" void kernel_launch(void* const* d_in, const int* in_sizes, int n_in,
                              void* d_out, int out_size, void* d_ws, size_t ws_size,
                              hipStream_t stream) {
  (void)in_sizes; (void)n_in; (void)out_size; (void)ws_size;
  const float* Xe   = (const float*)d_in[0];
  const float* Xt   = (const float*)d_in[1];
  const float* W1   = (const float*)d_in[2];
  const float* b1   = (const float*)d_in[3];
  const float* W2   = (const float*)d_in[4];
  const float* b2   = (const float*)d_in[5];
  const float* Wfc  = (const float*)d_in[6];
  const float* bfc  = (const float*)d_in[7];
  const float* Wih  = (const float*)d_in[8];
  const float* bih  = (const float*)d_in[9];
  const float* Whh  = (const float*)d_in[10];
  const float* bhh  = (const float*)d_in[11];
  const float* Wfin = (const float*)d_in[12];
  const float* bfin = (const float*)d_in[13];
  float* out = (float*)d_out;

  // Workspace layout (all L2-resident, ~4.6 MB total):
  char* ws = (char*)d_ws;
  float*    preEnc  = (float*)ws;                          // 8192*128 f32 = 4 MiB
  float*    h_g     = (float*)(ws + (size_t)(4u << 20));   // 128*128 f32
  float*    cc_g    = h_g   + BQ * HQ;                     // 128*128 f32
  float*    ctx_g   = cc_g  + BQ * HQ;                     // 128*128 f32
  float*    gates_g = ctx_g + BQ * HQ;                     // 128*512 f32
  _Float16* cc16_g  = (_Float16*)(gates_g + BQ * 4 * HQ);  // 128*128 f16

  init_state_kernel<<<(BQ * HQ + 255) / 256, 256, 0, stream>>>(h_g, cc_g, cc16_g);
  precompute_enc_kernel<<<(BQ * TQ) / 16, 256, 0, stream>>>(Xe, W1, b1, preEnc);
  decoder_scan_kernel<<<1, 1024, 0, stream>>>(Xe, Xt, W1, W2, b2, Wfc, bfc,
                                              Wih, bih, Whh, bhh, Wfin, bfin,
                                              preEnc, h_g, cc_g, cc16_g,
                                              ctx_g, gates_g, out);
}
